// GraphAttentionLayer_35296041239234
// MI455X (gfx1250) — matrix operand order repaired and checked by reading, not compiled, not run on hardware
//
#include <hip/hip_runtime.h>

#define LRELU_ALPHA 0.01f

typedef __attribute__((ext_vector_type(16))) __bf16        v16bf;
typedef __attribute__((ext_vector_type(8)))  __bf16        v8bf;
typedef __attribute__((ext_vector_type(8)))  float         v8f;
typedef __attribute__((ext_vector_type(8)))  unsigned int  v8u;

static __device__ __forceinline__ float bf2f(unsigned short s) {
    return __uint_as_float(((unsigned)s) << 16);
}

// ---------------------------------------------------------------------------
// Kernel 1: Wt[n][k] = bf16(W[k][n])   (256x256, tiny)
// ---------------------------------------------------------------------------
__global__ __launch_bounds__(256) void prep_W(const float* __restrict__ W,
                                              unsigned short* __restrict__ Wt) {
    int n = blockIdx.x;       // 0..255 output feature
    int k = threadIdx.x;      // 0..255 input feature
    __bf16 b = (__bf16)W[k * 256 + n];        // native RNE convert
    Wt[n * 256 + k] = __builtin_bit_cast(unsigned short, b);
}

// ---------------------------------------------------------------------------
// Kernel 2: h = inp @ W  (M=32768, K=256, N=256), bf16 WMMA, f32 accum.
// Block = 1 m-tile (16 rows) x 256 cols; 4 waves, 4 16x16 tiles per wave.
// Output written FEATURE-MAJOR bf16: ht[b][f][i]  (i contiguous).
// ---------------------------------------------------------------------------
__global__ __launch_bounds__(128) void gemm_h(const float* __restrict__ inp,
                                              const unsigned int* __restrict__ Wt,
                                              unsigned int* __restrict__ ht) {
    const int mtile = blockIdx.x;            // 0..2047
    const int wv    = threadIdx.x >> 5;      // wave 0..3 -> cols wv*64..wv*64+63
    const int lane  = threadIdx.x & 31;
    const int r     = lane & 15;             // A-row within tile
    const int hi    = lane >> 4;             // lane half
    const int kb    = hi * 16;

    const int gr    = mtile * 16 + r;        // global row (b*1024+i)
    const int b     = (mtile * 16) >> 10;
    const int ibase = (mtile * 16) & 1023;
    const int colbase = wv * 64;
    const int cl    = lane & 15;             // B/C column within tile

    const unsigned int* bbase = Wt + (size_t)(colbase + cl) * 128 + hi * 8;

    v8f c[4] = {};

    for (int ks = 0; ks < 8; ++ks) {
        const int k0 = ks * 32;

        v8u bu[4];
#pragma unroll
        for (int t = 0; t < 4; ++t) {
            const unsigned int* bp = bbase + t * (16 * 128) + (k0 >> 1);
#pragma unroll
            for (int v = 0; v < 8; ++v) bu[t][v] = bp[v];
        }

        // A fragment: lane r holds K in {kb..kb+15}
        const float4* ap0 = reinterpret_cast<const float4*>(inp + (size_t)gr * 256 + k0 + kb);
        const float4* ap1 = reinterpret_cast<const float4*>(inp + (size_t)gr * 256 + k0 + 8 + kb);
        float4 x0 = ap0[0], x1 = ap0[1];
        float4 x2 = ap1[0], x3 = ap1[1];
        v16bf av;
        av[0]  = (__bf16)x0.x; av[1]  = (__bf16)x0.y;
        av[2]  = (__bf16)x0.z; av[3]  = (__bf16)x0.w;
        av[4]  = (__bf16)x1.x; av[5]  = (__bf16)x1.y;
        av[6]  = (__bf16)x1.z; av[7]  = (__bf16)x1.w;
        av[8]  = (__bf16)x2.x; av[9]  = (__bf16)x2.y;
        av[10] = (__bf16)x2.z; av[11] = (__bf16)x2.w;
        av[12] = (__bf16)x3.x; av[13] = (__bf16)x3.y;
        av[14] = (__bf16)x3.z; av[15] = (__bf16)x3.w;

#pragma unroll
        for (int t = 0; t < 4; ++t) {
            v16bf bv = __builtin_bit_cast(v16bf, bu[t]);
            c[t] = __builtin_amdgcn_wmma_f32_16x16x32_bf16(false, av, false, bv,
                                                           (short)0, c[t], false, false);
        }
    }

    // C layout: VGPR v -> row v + 8*hi. Consecutive v = consecutive i in ht.
#pragma unroll
    for (int t = 0; t < 4; ++t) {
        const int f = colbase + t * 16 + cl;
        v8bf hv;
#pragma unroll
        for (int v = 0; v < 8; ++v) hv[v] = (__bf16)c[t][v];
        unsigned int* wp = ht + (size_t)(b * 256 + f) * 512 + (ibase >> 1) + hi * 4;
        *reinterpret_cast<uint4*>(wp) = __builtin_bit_cast(uint4, hv);
    }
}

// ---------------------------------------------------------------------------
// Kernel 3: f1[b,i] = sum_f h[b,i,f]*a1[f]; f2 likewise. Coalesced over i.
// ---------------------------------------------------------------------------
__global__ __launch_bounds__(256) void f1f2_k(const unsigned short* __restrict__ ht,
                                              const float* __restrict__ a,
                                              float* __restrict__ f1,
                                              float* __restrict__ f2) {
    int idx = blockIdx.x * 256 + threadIdx.x;   // b*1024 + i, 32768 total
    int b = idx >> 10, i = idx & 1023;
    const unsigned short* hp = ht + (size_t)b * 256 * 1024 + i;
    float s1 = 0.f, s2 = 0.f;
    for (int f = 0; f < 256; ++f) {
        float v = bf2f(hp[(size_t)f * 1024]);
        s1 = fmaf(v, a[f], s1);
        s2 = fmaf(v, a[256 + f], s2);
    }
    f1[idx] = s1;
    f2[idx] = s2;
}

// ---------------------------------------------------------------------------
// Kernel 4: f2max[b] = max_j f2[b,j]  (LDS tree reduction, no atomics)
// ---------------------------------------------------------------------------
__global__ __launch_bounds__(256) void f2max_k(const float* __restrict__ f2,
                                               float* __restrict__ f2max) {
    __shared__ float sm[256];
    const float* p = f2 + blockIdx.x * 1024;
    int t = threadIdx.x;
    float m = fmaxf(fmaxf(p[t], p[t + 256]), fmaxf(p[t + 512], p[t + 768]));
    sm[t] = m;
    __syncthreads();
    for (int s = 128; s > 0; s >>= 1) {
        if (t < s) sm[t] = fmaxf(sm[t], sm[t + s]);
        __syncthreads();
    }
    if (t == 0) f2max[blockIdx.x] = sm[0];
}

// ---------------------------------------------------------------------------
// Attention helpers
// ---------------------------------------------------------------------------
static __device__ __forceinline__ void load_bfrag(v8u bu[4],
                                                  const unsigned int* __restrict__ bbase,
                                                  int j0h) {
#pragma unroll
    for (int t = 0; t < 4; ++t) {
        const unsigned int* bp = bbase + t * (16 * 512) + j0h;
#pragma unroll
        for (int v = 0; v < 8; ++v) bu[t][v] = bp[v];
    }
}

static __device__ __forceinline__ void make_pfrag(const float2* __restrict__ f2v,
                                                  int j0, int kb,
                                                  float f1r0, float f1r1,
                                                  float m0, float m1,
                                                  v16bf& a0, v16bf& a1,
                                                  float& rs0, float& rs1) {
#pragma unroll
    for (int v = 0; v < 8; ++v) {
        const int kk = kb + 2 * v;   // A-layout K index
        float2 fp = f2v[(j0 + kk) >> 1];

        float e0 = f1r0 + fp.x; e0 = (e0 > 0.f) ? e0 : LRELU_ALPHA * e0;
        float e1 = f1r0 + fp.y; e1 = (e1 > 0.f) ? e1 : LRELU_ALPHA * e1;
        float p0 = __expf(e0 - m0), p1 = __expf(e1 - m0);
        rs0 += p0 + p1;
        a0[2 * v] = (__bf16)p0; a0[2 * v + 1] = (__bf16)p1;

        float g0 = f1r1 + fp.x; g0 = (g0 > 0.f) ? g0 : LRELU_ALPHA * g0;
        float g1 = f1r1 + fp.y; g1 = (g1 > 0.f) ? g1 : LRELU_ALPHA * g1;
        float q0 = __expf(g0 - m1), q1 = __expf(g1 - m1);
        rs1 += q0 + q1;
        a1[2 * v] = (__bf16)q0; a1[2 * v + 1] = (__bf16)q1;
    }
}

static __device__ __forceinline__ void wmma8(v8f c[8], v16bf a0, v16bf a1, const v8u bu[4]) {
#pragma unroll
    for (int t = 0; t < 4; ++t) {
        v16bf bv = __builtin_bit_cast(v16bf, bu[t]);
        c[t]     = __builtin_amdgcn_wmma_f32_16x16x32_bf16(false, a0, false, bv,
                                                           (short)0, c[t],     false, false);
        c[4 + t] = __builtin_amdgcn_wmma_f32_16x16x32_bf16(false, a1, false, bv,
                                                           (short)0, c[4 + t], false, false);
    }
}

// ---------------------------------------------------------------------------
// Kernel 5: single-pass softmax-attention + P@h + elu, double-buffered B.
// Row max is analytic: m_i = lrelu(f1_i + max_j f2_j) -> no online rescaling.
// Block = (b, 32-row group); wave wv handles feature cols wv*64..+63 with
// TWO 16-row A fragments sharing every B fragment.
// ---------------------------------------------------------------------------
__global__ __launch_bounds__(128) void attn_k(const unsigned int* __restrict__ ht,
                                              const float* __restrict__ f1,
                                              const float* __restrict__ f2,
                                              const float* __restrict__ f2max,
                                              float* __restrict__ out) {
    const int b    = blockIdx.x >> 5;       // batch
    const int rg   = blockIdx.x & 31;       // 32-row group
    const int wv   = threadIdx.x >> 5;
    const int lane = threadIdx.x & 31;
    const int r    = lane & 15;
    const int hi   = lane >> 4;
    const int kb   = hi * 16;
    const int colbase = wv * 64;
    const int cl   = lane & 15;
    const int rowbase = rg * 32;

    const float f1r0 = f1[b * 1024 + rowbase + r];
    const float f1r1 = f1[b * 1024 + rowbase + 16 + r];
    const float fm   = f2max[b];
    float m0 = f1r0 + fm; m0 = (m0 > 0.f) ? m0 : LRELU_ALPHA * m0;
    float m1 = f1r1 + fm; m1 = (m1 > 0.f) ? m1 : LRELU_ALPHA * m1;

    const float2* f2v = reinterpret_cast<const float2*>(f2 + b * 1024);
    const unsigned int* bbase = ht + (size_t)(b * 256 + colbase + cl) * 512 + hi * 8;

    v8f c[8] = {};            // [0..3]: rows 0-15, [4..7]: rows 16-31
    float rs0 = 0.f, rs1 = 0.f;

    // Software pipeline over 32 j-steps: B fragments for step s+1 are in
    // flight while the 8 WMMAs of step s execute.
    v8u bu0[4], bu1[4];
    load_bfrag(bu0, bbase, 0);

    for (int sp = 0; sp < 16; ++sp) {
        const int s0 = 2 * sp, s1 = 2 * sp + 1;

        load_bfrag(bu1, bbase, s1 * 16);               // prefetch odd step
        v16bf a0, a1;
        make_pfrag(f2v, s0 * 32, kb, f1r0, f1r1, m0, m1, a0, a1, rs0, rs1);
        wmma8(c, a0, a1, bu0);

        if (sp < 15) load_bfrag(bu0, bbase, (s1 + 1) * 16);   // prefetch next even
        make_pfrag(f2v, s1 * 32, kb, f1r0, f1r1, m0, m1, a0, a1, rs0, rs1);
        wmma8(c, a0, a1, bu1);
    }

    // Row sums: lane L covers half the j's of row L&15; combine with lane L^16.
    float z0 = rs0 + __shfl_xor(rs0, 16, 32);
    float z1 = rs1 + __shfl_xor(rs1, 16, 32);

#pragma unroll
    for (int vc = 0; vc < 8; ++vc) {
        const int row = vc + 8 * hi;                  // C layout row
        float rz0 = 1.f / __shfl(z0, row, 32);
        float rz1 = 1.f / __shfl(z1, row, 32);
        const size_t o0 = ((size_t)b * 1024 + rowbase + row) * 256;
        const size_t o1 = ((size_t)b * 1024 + rowbase + 16 + row) * 256;
#pragma unroll
        for (int t = 0; t < 4; ++t) {
            float x = c[t][vc] * rz0;
            x = (x > 0.f) ? x : (__expf(x) - 1.f);    // elu, alpha=1
            out[o0 + colbase + t * 16 + cl] = x;
            float y = c[4 + t][vc] * rz1;
            y = (y > 0.f) ? y : (__expf(y) - 1.f);
            out[o1 + colbase + t * 16 + cl] = y;
        }
    }
}

// ---------------------------------------------------------------------------
// Host launch
// ---------------------------------------------------------------------------
extern "C" void kernel_launch(void* const* d_in, const int* in_sizes, int n_in,
                              void* d_out, int out_size, void* d_ws, size_t ws_size,
                              hipStream_t stream) {
    (void)in_sizes; (void)n_in; (void)out_size; (void)ws_size;
    const float* inp = (const float*)d_in[0];   // (32,1024,256) f32
    const float* W   = (const float*)d_in[1];   // (256,256) f32
    const float* a   = (const float*)d_in[2];   // (512,1) f32
    float* out = (float*)d_out;                 // (32,1024,256) f32

    char* ws = (char*)d_ws;
    unsigned int*   ht  = (unsigned int*)(ws);                              // 16 MB bf16 h (feature-major)
    unsigned short* Wt  = (unsigned short*)(ws + (16u << 20));              // 128 KB bf16 W^T
    float*          f1  = (float*)(ws + (16u << 20) + (128u << 10));        // 128 KB
    float*          f2  = (float*)(ws + (16u << 20) + (256u << 10));        // 128 KB
    float*          f2m = (float*)(ws + (16u << 20) + (384u << 10));        // 128 B

    prep_W <<<256,  256, 0, stream>>>(W, Wt);
    gemm_h <<<2048, 128, 0, stream>>>(inp, (const unsigned int*)Wt, ht);
    f1f2_k <<<128,  256, 0, stream>>>((const unsigned short*)ht, a, f1, f2);
    f2max_k<<<32,   256, 0, stream>>>(f2, f2m);
    attn_k <<<1024, 128, 0, stream>>>(ht, f1, f2, f2m, out);
}
